// QMoPSSM_32143535243682
// MI455X (gfx1250) — matrix operand order repaired and checked by reading, not compiled, not run on hardware
//
#include <hip/hip_runtime.h>
#include <math.h>

// ---------------- dims ----------------
#define DD   4096
#define HH   2048
#define SS   32
#define CC   32
#define GG   8
#define EE   1024
#define QQ   128
#define AA   32
#define BB   16
#define TT   64
#define SCC  1024           // S*C
#define DGG  512            // D/G
#define H3   6144           // 3*H
#define KH   6656           // DG + 3H
#define OBSK 5248           // D+E+Q
#define OUTW 12288          // per (b,t) output row

typedef __bf16          bf16x16 __attribute__((ext_vector_type(16)));
typedef float           f32x8   __attribute__((ext_vector_type(8)));
typedef unsigned short  u16x8   __attribute__((ext_vector_type(8)));
typedef unsigned short  u16x16  __attribute__((ext_vector_type(16)));

__device__ __forceinline__ unsigned short f2bf(float f) {
    unsigned int u = __float_as_uint(f);
    unsigned int lsb = (u >> 16) & 1u;
    u += 0x7fffu + lsb;
    return (unsigned short)(u >> 16);
}
__device__ __forceinline__ float sigm(float x) { return 1.0f / (1.0f + __expf(-x)); }
__device__ __forceinline__ float u01(unsigned x) {
    x ^= x >> 17; x *= 0xed5ad4bbu; x ^= x >> 11; x *= 0xac4c1b51u;
    x ^= x >> 15; x *= 0x31848babu; x ^= x >> 14;
    return (float)(x >> 8) * (1.0f / 16777216.0f);
}

// ---------------- fp32 -> bf16 weight conversion ----------------
__global__ void qm_cvt(const float* __restrict__ s, unsigned short* __restrict__ d, long long n) {
    long long i = (long long)blockIdx.x * blockDim.x + threadIdx.x;
    long long st = (long long)gridDim.x * blockDim.x;
    for (; i < n; i += st) d[i] = f2bf(s[i]);
}

__global__ void qm_zero(float* __restrict__ p, long long n) {
    long long i = (long long)blockIdx.x * blockDim.x + threadIdx.x;
    long long st = (long long)gridDim.x * blockDim.x;
    for (; i < n; i += st) p[i] = 0.0f;
}

// ---------------- generic M=16 WMMA bf16 GEMM ----------------
// Y[z][m][n] = sum_k X[z][m][k] * W[z][k][n] + bias[z][n]
// block = 256 threads = 8 waves; each wave owns TWO 16-column tiles (32 cols),
// sharing one A fragment per K-chunk across both WMMAs.
__global__ void __launch_bounds__(256)
qm_gemm(const unsigned short* __restrict__ X, int ldx, long long xZ,
        const unsigned short* __restrict__ W, long long wZ,
        const float* __restrict__ bias, int biasZ,
        float* __restrict__ Y, int ldy, long long yZ,
        int K, int N)
{
    const int z = blockIdx.z;
    X += (long long)z * xZ;
    W += (long long)z * wZ;
    Y += (long long)z * yZ;
    const float* bz = bias ? bias + (long long)z * biasZ : nullptr;

    const int wave = threadIdx.x >> 5;
    const int lane = threadIdx.x & 31;
    const int n0 = blockIdx.x * 256 + wave * 32;     // two 16-col tiles
    if (n0 >= N) return;                             // wave-uniform

    const int m  = lane & 15;                        // A row / C column
    const int kb = (lane < 16) ? 0 : 8;              // A-fragment K sub-base

    const unsigned short* xrow = X + (long long)m * ldx + kb;
    const unsigned short* wrow = W + (long long)kb * N + n0 + m; // B col n0+m, K sub-base kb

    f32x8 acc0 = {};
    f32x8 acc1 = {};
    for (int k0 = 0; k0 < K; k0 += 32) {
        // A fragment: elements 0..7 = K kb..kb+7 ; 8..15 = K kb+16..kb+23
        u16x8 alo = *(const u16x8*)(xrow + k0);
        u16x8 ahi = *(const u16x8*)(xrow + k0 + 16);
        u16x16 au = __builtin_shufflevector(alo, ahi,
                        0,1,2,3,4,5,6,7,8,9,10,11,12,13,14,15);
        bf16x16 a = __builtin_bit_cast(bf16x16, au);
        // Two adjacent B fragments: lane holds column m, K same order as A
        const unsigned short* wk = wrow + (long long)k0 * N;
        u16x16 bu0, bu1;
        #pragma unroll
        for (int i = 0; i < 16; ++i) {
            const int kk = (i < 8) ? i : i + 8;
            bu0[i] = wk[(long long)kk * N];
            bu1[i] = wk[(long long)kk * N + 16];
        }
        bf16x16 b0 = __builtin_bit_cast(bf16x16, bu0);
        bf16x16 b1 = __builtin_bit_cast(bf16x16, bu1);
        if (k0 + 32 < K)
            __builtin_prefetch(wrow + (long long)(k0 + 32) * N, 0, 3); // WGP-scope prefetch
        acc0 = __builtin_amdgcn_wmma_f32_16x16x32_bf16(
                   false, a, false, b0, (short)0, acc0, false, false);
        acc1 = __builtin_amdgcn_wmma_f32_16x16x32_bf16(
                   false, a, false, b1, (short)0, acc1, false, false);
    }

    const int mb = (lane < 16) ? 0 : 8;              // C rows mb..mb+7 in acc[0..7]
    const float bv0 = bz ? bz[n0 + m] : 0.0f;
    const float bv1 = bz ? bz[n0 + 16 + m] : 0.0f;
    float* yp = Y + (long long)mb * ldy + n0 + m;
    #pragma unroll
    for (int i = 0; i < 8; ++i) {
        yp[(long long)i * ldy]      = acc0[i] + bv0;
        yp[(long long)i * ldy + 16] = acc1[i] + bv1;
    }
}

// ---------------- per-step elementwise kernels ----------------
// mask states by reset, normalize action, stage bf16 inputs
__global__ void qm_prep(int t,
                        const float* __restrict__ action, const float* __restrict__ query,
                        const unsigned char* __restrict__ reset,
                        float* __restrict__ deter, unsigned short* __restrict__ deter_bf,
                        float* __restrict__ stoch, unsigned short* __restrict__ stoch_bf,
                        float* __restrict__ pdeter, unsigned short* __restrict__ pdeter_bf,
                        float* __restrict__ pstoch, unsigned short* __restrict__ pstoch_bf,
                        unsigned short* __restrict__ act_bf, unsigned short* __restrict__ qry_bf)
{
    int tid = blockIdx.x * blockDim.x + threadIdx.x;
    if (tid >= BB * DD) return;
    int b = tid / DD, d = tid % DD;
    float msk = reset[b * TT + t] ? 0.0f : 1.0f;
    float dv = deter[tid] * msk;   deter[tid] = dv;   deter_bf[tid] = f2bf(dv);
    float pv = pdeter[tid] * msk;  pdeter[tid] = pv;  pdeter_bf[tid] = f2bf(pv);
    if (d < SCC) {
        int i = b * SCC + d;
        float sv = stoch[i] * msk;   stoch[i] = sv;   stoch_bf[i] = f2bf(sv);
        float ps = pstoch[i] * msk;  pstoch[i] = ps;  pstoch_bf[i] = f2bf(ps);
    }
    if (d < AA) {
        float a = action[((long long)b * TT + t) * AA + d] * msk;
        a = a / fmaxf(1.0f, fabsf(a));
        act_bf[b * AA + d] = f2bf(a);
    }
    if (d < QQ) {
        float q = query[((long long)b * TT + t) * QQ + d] * msk;
        qry_bf[b * QQ + d] = f2bf(q);
    }
}

// build bf16 concat [g][m][KH] = [ deter_g(512) | xcat(6144) ]
__global__ void qm_build_hid(const float* __restrict__ deter, const float* __restrict__ xcat,
                             unsigned short* __restrict__ o)
{
    int tid = blockIdx.x * blockDim.x + threadIdx.x;
    if (tid >= GG * BB * KH) return;
    int g = tid / (BB * KH);
    int r = tid % (BB * KH);
    int m = r / KH, c = r % KH;
    float v = (c < DGG) ? deter[m * DD + g * DGG + c]
                        : xcat[m * H3 + (c - DGG)];
    o[tid] = f2bf(v);
}

// RMSNorm + SiLU -> bf16; one block per row
__global__ void qm_rms_silu(const float* __restrict__ x, const float* __restrict__ scale,
                            unsigned short* __restrict__ o, int N)
{
    int m = blockIdx.x;
    __shared__ float red[8];
    __shared__ float rtot;
    float ss = 0.0f;
    for (int i = threadIdx.x; i < N; i += 256) { float v = x[m * N + i]; ss += v * v; }
    #pragma unroll
    for (int off = 16; off; off >>= 1) ss += __shfl_xor(ss, off, 32);
    if ((threadIdx.x & 31) == 0) red[threadIdx.x >> 5] = ss;
    __syncthreads();
    if (threadIdx.x == 0) {
        float tsum = 0.0f;
        #pragma unroll
        for (int i = 0; i < 8; ++i) tsum += red[i];
        rtot = rsqrtf(tsum / (float)N + 1e-6f);
    }
    __syncthreads();
    float r = rtot;
    for (int i = threadIdx.x; i < N; i += 256) {
        float v = x[m * N + i] * r * scale[i];
        v = v * sigm(v);
        o[m * N + i] = f2bf(v);
    }
}

// GRU combine: deter = u*tanh(r*cand) + (1-u)*deter ; also writes d_out slice
__global__ void qm_gru(const float* __restrict__ Y, float* __restrict__ deter,
                       unsigned short* __restrict__ deter_bf,
                       float* __restrict__ out, int outOff, int t)
{
    int tid = blockIdx.x * blockDim.x + threadIdx.x;
    if (tid >= BB * DD) return;
    int m = tid / DD, d = tid % DD;
    int g = d >> 9, c = d & 511;
    const float* yb = Y + (long long)m * (GG * 3 * DGG) + g * (3 * DGG);
    float rg = sigm(yb[c]);
    float cd = yb[DGG + c];
    float up = sigm(yb[2 * DGG + c] - 1.0f);
    cd = tanhf(rg * cd);
    float nd = up * cd + (1.0f - up) * deter[tid];
    deter[tid] = nd;
    deter_bf[tid] = f2bf(nd);
    out[((long long)m * TT + t) * OUTW + outOff + d] = nd;
}

// build bf16 [m][OBSK] = [ deter(4096) | input_emb(1024) | query(128) ]
__global__ void qm_build_obs(const float* __restrict__ deter, const float* __restrict__ emb,
                             const unsigned short* __restrict__ qry_bf,
                             unsigned short* __restrict__ o, int t)
{
    int tid = blockIdx.x * blockDim.x + threadIdx.x;
    if (tid >= BB * OBSK) return;
    int m = tid / OBSK, c = tid % OBSK;
    if (c < DD)            o[tid] = f2bf(deter[m * DD + c]);
    else if (c < DD + EE)  o[tid] = f2bf(emb[((long long)m * TT + t) * EE + (c - DD)]);
    else                   o[tid] = qry_bf[m * QQ + (c - DD - EE)];
}

// softmax+unimix+categorical sample per (b,s); two waves: stoch / pstoch
__global__ void qm_sample(int t,
                          const float* __restrict__ logitbuf, const float* __restrict__ plogitbuf,
                          float* __restrict__ stoch, unsigned short* __restrict__ stoch_bf,
                          float* __restrict__ pstoch, unsigned short* __restrict__ pstoch_bf,
                          float* __restrict__ out)
{
    int bs = blockIdx.x;
    int b = bs / SS, s = bs % SS;
    int which = threadIdx.x >> 5;
    int lane  = threadIdx.x & 31;
    const float* L = which ? plogitbuf : logitbuf;
    float lg = L[b * SCC + s * CC + lane];
    float mx = lg;
    #pragma unroll
    for (int off = 16; off; off >>= 1) mx = fmaxf(mx, __shfl_xor(mx, off, 32));
    float e = __expf(lg - mx);
    float sum = e;
    #pragma unroll
    for (int off = 16; off; off >>= 1) sum += __shfl_xor(sum, off, 32);
    float p = 0.99f * (e / sum) + 0.01f / (float)CC;
    // inclusive prefix sum
    float cdf = p;
    #pragma unroll
    for (int off = 1; off < 32; off <<= 1) {
        float v = __shfl_up(cdf, off, 32);
        if (lane >= off) cdf += v;
    }
    unsigned seed = ((unsigned)t * 2654435761u) ^ ((unsigned)(b * SS + s) * 40503u)
                    ^ ((unsigned)which * 0x9e3779b9u) ^ 0xC0FFEEu;
    float r = u01(seed);
    unsigned long long bal = __ballot(cdf >= r);
    int idx = (bal == 0ull) ? (CC - 1) : (__ffsll(bal) - 1);
    float oh = (lane == idx) ? 1.0f : 0.0f;

    int si = b * SCC + s * CC + lane;
    if (which) { pstoch[si] = oh; pstoch_bf[si] = f2bf(oh); }
    else       { stoch[si]  = oh; stoch_bf[si]  = f2bf(oh); }
    long long ob = ((long long)b * TT + t) * OUTW;
    out[ob + (which ? 9216  : 4096)  + s * CC + lane] = oh;
    out[ob + (which ? 11264 : 10240) + s * CC + lane] = lg;
}

// ---------------- host ----------------
static inline size_t alignup(size_t x) { return (x + 255) & ~(size_t)255; }

extern "C" void kernel_launch(void* const* d_in, const int* in_sizes, int n_in,
                              void* d_out, int out_size, void* d_ws, size_t ws_size,
                              hipStream_t stream)
{
    (void)in_sizes; (void)n_in; (void)out_size; (void)ws_size;
    // inputs
    const float* input_emb = (const float*)d_in[0];
    const float* action    = (const float*)d_in[1];
    const float* query     = (const float*)d_in[2];
    const unsigned char* reset = (const unsigned char*)d_in[3];
    // params (jax pytree alphabetical flatten)
    const float* c_grub   = (const float*)d_in[4];
    const float* c_gruw   = (const float*)d_in[5];
    const float* c_hid0b  = (const float*)d_in[6];
    const float* c_hid0w  = (const float*)d_in[7];
    const float* c_hnorm  = (const float*)d_in[8];
    const float* c_in0b   = (const float*)d_in[9];
    const float* c_in0w   = (const float*)d_in[10];
    const float* c_in1b   = (const float*)d_in[11];
    const float* c_in1w   = (const float*)d_in[12];
    const float* c_in2b   = (const float*)d_in[13];
    const float* c_in2w   = (const float*)d_in[14];
    const float* obs0b    = (const float*)d_in[15];
    const float* obs0w    = (const float*)d_in[16];
    const float* obs0n    = (const float*)d_in[17];
    const float* logitb   = (const float*)d_in[18];
    const float* logitw   = (const float*)d_in[19];
    const float* pobs0b   = (const float*)d_in[22];
    const float* pobs0w   = (const float*)d_in[23];
    const float* pobs0n   = (const float*)d_in[24];
    const float* plogitb  = (const float*)d_in[25];
    const float* plogitw  = (const float*)d_in[26];
    const float* p_grub   = (const float*)d_in[27];
    const float* p_gruw   = (const float*)d_in[28];
    const float* p_hid0b  = (const float*)d_in[29];
    const float* p_hid0w  = (const float*)d_in[30];
    const float* p_hnorm  = (const float*)d_in[31];
    const float* p_in0b   = (const float*)d_in[32];
    const float* p_in0w   = (const float*)d_in[33];
    const float* p_in1b   = (const float*)d_in[34];
    const float* p_in1w   = (const float*)d_in[35];
    const float* p_in2b   = (const float*)d_in[36];
    const float* p_in2w   = (const float*)d_in[37];

    float* out = (float*)d_out;

    // ---- workspace layout ----
    char* ws = (char*)d_ws;
    size_t cur = 0;
    auto ubuf = [&](size_t elems) {  // bf16 / u16
        unsigned short* p = (unsigned short*)(ws + cur);
        cur = alignup(cur + elems * 2); return p;
    };
    auto fbuf = [&](size_t elems) {
        float* p = (float*)(ws + cur);
        cur = alignup(cur + elems * 4); return p;
    };
    // bf16 weights
    unsigned short* w_c_in0  = ubuf((size_t)DD * HH);
    unsigned short* w_c_in1  = ubuf((size_t)SCC * HH);
    unsigned short* w_c_in2  = ubuf((size_t)AA * HH);
    unsigned short* w_c_hid  = ubuf((size_t)GG * KH * DGG);
    unsigned short* w_c_gru  = ubuf((size_t)GG * DGG * 3 * DGG);
    unsigned short* w_p_in0  = ubuf((size_t)DD * HH);
    unsigned short* w_p_in1  = ubuf((size_t)SCC * HH);
    unsigned short* w_p_in2  = ubuf((size_t)QQ * HH);
    unsigned short* w_p_hid  = ubuf((size_t)GG * KH * DGG);
    unsigned short* w_p_gru  = ubuf((size_t)GG * DGG * 3 * DGG);
    unsigned short* w_obs0   = ubuf((size_t)OBSK * HH);
    unsigned short* w_pobs0  = ubuf((size_t)OBSK * HH);
    unsigned short* w_logit  = ubuf((size_t)HH * SCC);
    unsigned short* w_plogit = ubuf((size_t)HH * SCC);
    // fp32 states (zeroed each launch)
    float* deter  = fbuf(BB * DD);
    float* stoch  = fbuf(BB * SCC);
    float* pdeter = fbuf(BB * DD);
    float* pstoch = fbuf(BB * SCC);
    // bf16 staging
    unsigned short* deter_bf  = ubuf(BB * DD);
    unsigned short* stoch_bf  = ubuf(BB * SCC);
    unsigned short* pdeter_bf = ubuf(BB * DD);
    unsigned short* pstoch_bf = ubuf(BB * SCC);
    unsigned short* act_bf    = ubuf(BB * AA);
    unsigned short* qry_bf    = ubuf(BB * QQ);
    // activations
    float* xcat   = fbuf(BB * H3);
    float* pxcat  = fbuf(BB * H3);
    unsigned short* hidin  = ubuf((size_t)GG * BB * KH);
    unsigned short* phidin = ubuf((size_t)GG * BB * KH);
    float* hid    = fbuf(BB * DD);
    float* phid   = fbuf(BB * DD);
    unsigned short* hidn_bf  = ubuf(BB * DD);
    unsigned short* phidn_bf = ubuf(BB * DD);
    float* gruo   = fbuf((size_t)BB * GG * 3 * DGG);
    float* pgruo  = fbuf((size_t)BB * GG * 3 * DGG);
    unsigned short* obsin  = ubuf((size_t)BB * OBSK);
    unsigned short* pobsin = ubuf((size_t)BB * OBSK);
    float* obsx   = fbuf(BB * HH);
    float* pobsx  = fbuf(BB * HH);
    unsigned short* obsxn_bf  = ubuf(BB * HH);
    unsigned short* pobsxn_bf = ubuf(BB * HH);
    float* lgt    = fbuf(BB * SCC);
    float* plgt   = fbuf(BB * SCC);

    // ---- one-time: convert weights fp32 -> bf16 ----
    auto cvt = [&](const float* s, unsigned short* d, size_t n) {
        int blocks = (int)((n + 255) / 256); if (blocks > 8192) blocks = 8192;
        qm_cvt<<<blocks, 256, 0, stream>>>(s, d, (long long)n);
    };
    cvt(c_in0w,  w_c_in0,  (size_t)DD * HH);
    cvt(c_in1w,  w_c_in1,  (size_t)SCC * HH);
    cvt(c_in2w,  w_c_in2,  (size_t)AA * HH);
    cvt(c_hid0w, w_c_hid,  (size_t)GG * KH * DGG);
    cvt(c_gruw,  w_c_gru,  (size_t)GG * DGG * 3 * DGG);
    cvt(p_in0w,  w_p_in0,  (size_t)DD * HH);
    cvt(p_in1w,  w_p_in1,  (size_t)SCC * HH);
    cvt(p_in2w,  w_p_in2,  (size_t)QQ * HH);
    cvt(p_hid0w, w_p_hid,  (size_t)GG * KH * DGG);
    cvt(p_gruw,  w_p_gru,  (size_t)GG * DGG * 3 * DGG);
    cvt(obs0w,   w_obs0,   (size_t)OBSK * HH);
    cvt(pobs0w,  w_pobs0,  (size_t)OBSK * HH);
    cvt(logitw,  w_logit,  (size_t)HH * SCC);
    cvt(plogitw, w_plogit, (size_t)HH * SCC);

    // zero recurrent states (deter..pstoch are contiguous)
    qm_zero<<<640, 256, 0, stream>>>(deter, (long long)(2 * (BB * DD + BB * SCC)));

    auto gemm = [&](const unsigned short* X, int ldx, long long xZ,
                    const unsigned short* W, long long wZ,
                    const float* bias, int biasZ,
                    float* Y, int ldy, long long yZ,
                    int K, int N, int Gz) {
        dim3 grid((N + 255) / 256, 1, Gz);
        qm_gemm<<<grid, 256, 0, stream>>>(X, ldx, xZ, W, wZ, bias, biasZ, Y, ldy, yZ, K, N);
    };

    const int EW = 256;
    for (int t = 0; t < TT; ++t) {
        qm_prep<<<(BB * DD + EW - 1) / EW, EW, 0, stream>>>(
            t, action, query, reset,
            deter, deter_bf, stoch, stoch_bf, pdeter, pdeter_bf, pstoch, pstoch_bf,
            act_bf, qry_bf);

        // core inputs: xcat = [in0(deter) | in1(stoch) | in2(action)]
        gemm(deter_bf, DD, 0, w_c_in0, 0, c_in0b, 0, xcat,        H3, 0, DD,  HH, 1);
        gemm(stoch_bf, SCC, 0, w_c_in1, 0, c_in1b, 0, xcat + HH,  H3, 0, SCC, HH, 1);
        gemm(act_bf,   AA,  0, w_c_in2, 0, c_in2b, 0, xcat + 2*HH,H3, 0, AA,  HH, 1);
        gemm(pdeter_bf, DD, 0, w_p_in0, 0, p_in0b, 0, pxcat,       H3, 0, DD,  HH, 1);
        gemm(pstoch_bf, SCC,0, w_p_in1, 0, p_in1b, 0, pxcat + HH,  H3, 0, SCC, HH, 1);
        gemm(qry_bf,    QQ, 0, w_p_in2, 0, p_in2b, 0, pxcat + 2*HH,H3, 0, QQ,  HH, 1);

        // hid0 block-diagonal
        qm_build_hid<<<(GG * BB * KH + EW - 1) / EW, EW, 0, stream>>>(deter,  xcat,  hidin);
        qm_build_hid<<<(GG * BB * KH + EW - 1) / EW, EW, 0, stream>>>(pdeter, pxcat, phidin);
        gemm(hidin,  KH, (long long)BB * KH, w_c_hid, (long long)KH * DGG,
             c_hid0b, DGG, hid,  DD, DGG, KH, DGG, GG);
        gemm(phidin, KH, (long long)BB * KH, w_p_hid, (long long)KH * DGG,
             p_hid0b, DGG, phid, DD, DGG, KH, DGG, GG);

        qm_rms_silu<<<BB, 256, 0, stream>>>(hid,  c_hnorm, hidn_bf,  DD);
        qm_rms_silu<<<BB, 256, 0, stream>>>(phid, p_hnorm, phidn_bf, DD);

        // gru block-diagonal: [16,512] x [512,1536] per group
        gemm(hidn_bf,  DD, DGG, w_c_gru, (long long)DGG * 3 * DGG,
             c_grub, 3 * DGG, gruo,  GG * 3 * DGG, 3 * DGG, DGG, 3 * DGG, GG);
        gemm(phidn_bf, DD, DGG, w_p_gru, (long long)DGG * 3 * DGG,
             p_grub, 3 * DGG, pgruo, GG * 3 * DGG, 3 * DGG, DGG, 3 * DGG, GG);

        qm_gru<<<(BB * DD + EW - 1) / EW, EW, 0, stream>>>(gruo,  deter,  deter_bf,  out, 0,    t);
        qm_gru<<<(BB * DD + EW - 1) / EW, EW, 0, stream>>>(pgruo, pdeter, pdeter_bf, out, 5120, t);

        // obs paths
        qm_build_obs<<<(BB * OBSK + EW - 1) / EW, EW, 0, stream>>>(deter,  input_emb, qry_bf, obsin,  t);
        qm_build_obs<<<(BB * OBSK + EW - 1) / EW, EW, 0, stream>>>(pdeter, input_emb, qry_bf, pobsin, t);
        gemm(obsin,  OBSK, 0, w_obs0,  0, obs0b,  0, obsx,  HH, 0, OBSK, HH, 1);
        gemm(pobsin, OBSK, 0, w_pobs0, 0, pobs0b, 0, pobsx, HH, 0, OBSK, HH, 1);

        qm_rms_silu<<<BB, 256, 0, stream>>>(obsx,  obs0n,  obsxn_bf,  HH);
        qm_rms_silu<<<BB, 256, 0, stream>>>(pobsx, pobs0n, pobsxn_bf, HH);

        gemm(obsxn_bf,  HH, 0, w_logit,  0, logitb,  0, lgt,  SCC, 0, HH, SCC, 1);
        gemm(pobsxn_bf, HH, 0, w_plogit, 0, plogitb, 0, plgt, SCC, 0, HH, SCC, 1);
        // MIX==1 => pmix softmax == 1.0, plogit == pobslogit output exactly.

        qm_sample<<<BB * SS, 64, 0, stream>>>(t, lgt, plgt,
                                              stoch, stoch_bf, pstoch, pstoch_bf, out);
    }
}
